// TSModel_42545946034852
// MI455X (gfx1250) — matrix operand order repaired and checked by reading, not compiled, not run on hardware
//
#include <hip/hip_runtime.h>
#include <hip/hip_bf16.h>

// ---------------------------------------------------------------------------
// 4-layer BERT encoder forward on gfx1250: bf16 WMMA (f32 accumulate),
// double-buffered async global->LDS staging (GLOBAL_LOAD_ASYNC_TO_LDS_B64,
// software-pipelined on ASYNCcnt's in-order completion), bf16 dataflow.
// B=16, S=512, D=1024, H=16 (d=64), F=4096, M = B*S = 8192.
// ---------------------------------------------------------------------------

typedef __attribute__((ext_vector_type(16))) __bf16 v16bf;
typedef __attribute__((ext_vector_type(8)))  float  v8f;
typedef unsigned short ushort_t;

__device__ __forceinline__ unsigned short f2bf(float f) {
    union { float f; unsigned u; } x; x.f = f;
    unsigned r = x.u + 0x7FFFu + ((x.u >> 16) & 1u);   // round to nearest even
    return (unsigned short)(r >> 16);
}

union FragBF { v16bf v; unsigned u[8]; };

__device__ __forceinline__ v8f wmma_bf16(const v16bf& a, const v16bf& b, v8f c) {
    return __builtin_amdgcn_wmma_f32_16x16x32_bf16(
        false, a, false, b, (short)0, c, false, false);
}

// A-fragment K index for 16-bit A 16x32 (ISA 7.12.2).
__device__ __forceinline__ int a_kk(int j, int half) {
    return ((j < 4) ? (j * 2) : (16 + (j - 4) * 2)) + half * 8;
}

// Async copy of 8 bytes global -> LDS (per-lane addresses), ASYNCcnt-tracked.
__device__ __forceinline__ void async_b64(const ushort_t* sbase, unsigned voff_bytes,
                                          ushort_t* lds_ptr) {
    const unsigned lds = (unsigned)(unsigned long long)lds_ptr;  // low 32 = LDS addr
    asm volatile("global_load_async_to_lds_b64 %0, %1, %2"
                 :: "v"(lds), "v"(voff_bytes), "s"(sbase) : "memory");
}
// Wait until <= N async ops outstanding (in-order completion).
__device__ __forceinline__ void wait_async0() {
    asm volatile("s_wait_asynccnt 0" ::: "memory");
}
__device__ __forceinline__ void wait_async8() {
    asm volatile("s_wait_asynccnt 8" ::: "memory");
}

// ---------------------------------------------------------------------------
// Weight transpose + convert: W[K,N] fp32 -> Wt[N,K] bf16.  32x32 LDS tile.
// ---------------------------------------------------------------------------
__global__ __launch_bounds__(256) void transpose_f32_to_bf16(
    const float* __restrict__ W, ushort_t* __restrict__ Wt, int K, int N)
{
    __shared__ ushort_t Ls[32][33];
    const int k0 = blockIdx.y * 32, n0 = blockIdx.x * 32;
    const int tid = threadIdx.x;
    const int r = tid >> 3;           // 0..31
    const int c = (tid & 7) * 4;      // 0..28
    const float4 f = *(const float4*)(W + (size_t)(k0 + r) * N + n0 + c);
    Ls[r][c + 0] = f2bf(f.x); Ls[r][c + 1] = f2bf(f.y);
    Ls[r][c + 2] = f2bf(f.z); Ls[r][c + 3] = f2bf(f.w);
    __syncthreads();
    const unsigned a0 = Ls[c + 0][r], a1 = Ls[c + 1][r];
    const unsigned a2 = Ls[c + 2][r], a3 = Ls[c + 3][r];
    uint2 o; o.x = a0 | (a1 << 16); o.y = a2 | (a3 << 16);
    *(uint2*)(Wt + (size_t)(n0 + r) * K + k0 + c) = o;
}

// Elementwise fp32 -> bf16 (4 elems/thread).
__global__ __launch_bounds__(256) void f32_to_bf16_k(
    const float* __restrict__ in, ushort_t* __restrict__ out)
{
    const size_t i = ((size_t)blockIdx.x * 256 + threadIdx.x) * 4;
    const float4 f = *(const float4*)(in + i);
    uint2 o;
    o.x = (unsigned)f2bf(f.x) | ((unsigned)f2bf(f.y) << 16);
    o.y = (unsigned)f2bf(f.z) | ((unsigned)f2bf(f.w) << 16);
    *(uint2*)(out + i) = o;
}

// ---------------------------------------------------------------------------
// GEMM: out[M,N] = A[M,K](bf16) @ Wt[N,K](bf16)^T + bias (+ReLU | +resid)
// Block 128x128xK32, 256 thr = 8 waves (4x2), wave tile 32x64 (8 WMMA).
// Double-buffered LDS; next tile's 8 asyncs stay in flight behind the WMMAs
// (steady-state wait is s_wait_asynccnt 8; completion is in-order).
// EPI: 0 = bias -> bf16, 1 = bias+relu -> bf16, 2 = bias+resid -> fp32
// ---------------------------------------------------------------------------
template <int EPI>
__global__ __launch_bounds__(256) void gemm_bf16_wmma(
    const ushort_t* __restrict__ A, const ushort_t* __restrict__ Wt,
    const float* __restrict__ bias, const float* __restrict__ resid,
    void* __restrict__ outp, int M, int N, int K)
{
    constexpr int BM = 128, BN = 128, BK = 32;
    constexpr int LDT = 36;                     // halves; 72B rows (8B aligned)

    __shared__ ushort_t As[2][BM * LDT];
    __shared__ ushort_t Bs[2][BN * LDT];

    const int tid  = threadIdx.x;
    const int lane = tid & 31;
    const int wid  = tid >> 5;
    const int wm   = wid >> 1;
    const int wn   = wid & 1;
    const int half = lane >> 4;
    const int l16  = lane & 15;
    const int bm0  = blockIdx.y * BM;
    const int bn0  = blockIdx.x * BN;

    const ushort_t* Ab = A  + (size_t)bm0 * K;
    const ushort_t* Bb = Wt + (size_t)bn0 * K;

    const int srow = tid >> 3, scol = (tid & 7) * 4;   // staging: 4 passes x 32 rows

    const v8f zf = {0.f, 0.f, 0.f, 0.f, 0.f, 0.f, 0.f, 0.f};
    v8f acc[2][4];
#pragma unroll
    for (int mi = 0; mi < 2; ++mi)
#pragma unroll
        for (int ni = 0; ni < 4; ++ni) acc[mi][ni] = zf;

    // 8 async instructions per wave per tile (4 for A, 4 for B).
    auto stage = [&](int k0, int buf) {
#pragma unroll
        for (int p = 0; p < 4; ++p) {
            const int r = srow + p * 32;
            async_b64(Ab, (unsigned)(((size_t)r * K + k0 + scol) * 2),
                      &As[buf][r * LDT + scol]);
        }
#pragma unroll
        for (int p = 0; p < 4; ++p) {
            const int r = srow + p * 32;
            async_b64(Bb, (unsigned)(((size_t)r * K + k0 + scol) * 2),
                      &Bs[buf][r * LDT + scol]);
        }
    };

    const int NT = K / BK;
    stage(0, 0);                                 // prologue: tile 0 in flight

    for (int t = 0; t < NT; ++t) {
        const int buf = t & 1;
        if (t + 1 < NT) {
            stage((t + 1) * BK, buf ^ 1);        // prefetch next tile
            wait_async8();                       // current tile landed; next in flight
        } else {
            wait_async0();
        }
        __syncthreads();

        FragBF a[2];
#pragma unroll
        for (int mi = 0; mi < 2; ++mi) {
            const int m = wm * 32 + mi * 16 + l16;
#pragma unroll
            for (int j = 0; j < 8; ++j)
                a[mi].u[j] = *(const unsigned*)&As[buf][m * LDT + a_kk(j, half)];
        }
#pragma unroll
        for (int ni = 0; ni < 4; ++ni) {
            FragBF b;
            const int n = wn * 64 + ni * 16 + l16;
#pragma unroll
            for (int j = 0; j < 8; ++j)
                b.u[j] = *(const unsigned*)&Bs[buf][n * LDT + a_kk(j, half)];
#pragma unroll
            for (int mi = 0; mi < 2; ++mi)
                acc[mi][ni] = wmma_bf16(a[mi].v, b.v, acc[mi][ni]);
        }
        __syncthreads();                         // readers done before buf reuse
    }

#pragma unroll
    for (int mi = 0; mi < 2; ++mi) {
#pragma unroll
        for (int ni = 0; ni < 4; ++ni) {
            const int col = bn0 + wn * 64 + ni * 16 + l16;
            const float bv = bias[col];
#pragma unroll
            for (int r = 0; r < 8; ++r) {
                const int row = bm0 + wm * 32 + mi * 16 + r + half * 8;
                float v = acc[mi][ni][r] + bv;
                if (EPI == 1) v = fmaxf(v, 0.f);
                if (EPI == 2) {
                    v += resid[(size_t)row * N + col];
                    ((float*)outp)[(size_t)row * N + col] = v;
                } else {
                    ((ushort_t*)outp)[(size_t)row * N + col] = f2bf(v);
                }
            }
        }
    }
}

// ---------------------------------------------------------------------------
// Fused attention (flash-style): bf16 Q/K/V in, bf16 ctx out.
// One block per (b, h, 64-query tile); 4 waves x 16 q-rows.
// ---------------------------------------------------------------------------
__global__ __launch_bounds__(128) void attn_wmma(
    const ushort_t* __restrict__ Q, const ushort_t* __restrict__ K,
    const ushort_t* __restrict__ V, ushort_t* __restrict__ O)
{
    constexpr int S = 512, D = 1024, HD = 64, LDK = 68;  // 136B rows (8B aligned)
    const int qb = blockIdx.x & 7;
    const int h  = (blockIdx.x >> 3) & 15;
    const int b  = blockIdx.x >> 7;

    const int tid  = threadIdx.x;
    const int lane = tid & 31;
    const int wid  = tid >> 5;
    const int half = lane >> 4;
    const int l16  = lane & 15;

    __shared__ ushort_t Ksh[64 * LDK];       // [key][d]
    __shared__ ushort_t Vsh[64 * LDK];       // [d][key]
    __shared__ ushort_t Psh[4][16 * LDK];    // per-wave P staging

    const ushort_t* Qp = Q + (size_t)(b * S + qb * 64 + wid * 16 + l16) * D + h * HD;
    FragBF aq[2];
#pragma unroll
    for (int f = 0; f < 2; ++f)
#pragma unroll
        for (int j = 0; j < 8; ++j)
            aq[f].u[j] = *(const unsigned*)&Qp[f * 32 + a_kk(j, half)];

    const v8f zf = {0.f, 0.f, 0.f, 0.f, 0.f, 0.f, 0.f, 0.f};
    v8f acc_o[4]; float mrow[8], lrow[8];
#pragma unroll
    for (int di = 0; di < 4; ++di) acc_o[di] = zf;
#pragma unroll
    for (int r = 0; r < 8; ++r) { mrow[r] = -3.4e38f; lrow[r] = 0.f; }

    for (int kb = 0; kb < S / 64; ++kb) {
        const ushort_t* Kb = K + (size_t)(b * S + kb * 64) * D + h * HD;
        const ushort_t* Vb = V + (size_t)(b * S + kb * 64) * D + h * HD;
#pragma unroll
        for (int p = 0; p < 8; ++p) {
            const int kr = (tid >> 4) + p * 8;
            const int cv = (tid & 15) * 4;
            async_b64(Kb, (unsigned)(((size_t)kr * D + cv) * 2), &Ksh[kr * LDK + cv]);
            const uint2 vv = *(const uint2*)(Vb + (size_t)kr * D + cv);
            Vsh[(cv + 0) * LDK + kr] = (ushort_t)(vv.x);
            Vsh[(cv + 1) * LDK + kr] = (ushort_t)(vv.x >> 16);
            Vsh[(cv + 2) * LDK + kr] = (ushort_t)(vv.y);
            Vsh[(cv + 3) * LDK + kr] = (ushort_t)(vv.y >> 16);
        }
        wait_async0();
        __syncthreads();

        // S = Q @ K^T * scale
        v8f sacc[4];
#pragma unroll
        for (int ni = 0; ni < 4; ++ni) {
            sacc[ni] = zf;
            FragBF bk0, bk1;
            const int key = ni * 16 + l16;
#pragma unroll
            for (int j = 0; j < 8; ++j) {
                const int kk = a_kk(j, half);
                bk0.u[j] = *(const unsigned*)&Ksh[key * LDK + kk];
                bk1.u[j] = *(const unsigned*)&Ksh[key * LDK + 32 + kk];
            }
            sacc[ni] = wmma_bf16(aq[0].v, bk0.v, sacc[ni]);
            sacc[ni] = wmma_bf16(aq[1].v, bk1.v, sacc[ni]);
        }

        // Online softmax (row = r + 8*half spread over 16 lanes).
#pragma unroll
        for (int r = 0; r < 8; ++r) {
            float mx = -3.4e38f;
#pragma unroll
            for (int ni = 0; ni < 4; ++ni) {
                sacc[ni][r] *= 0.125f;
                mx = fmaxf(mx, sacc[ni][r]);
            }
            mx = fmaxf(mx, __shfl_xor(mx, 1, 32));
            mx = fmaxf(mx, __shfl_xor(mx, 2, 32));
            mx = fmaxf(mx, __shfl_xor(mx, 4, 32));
            mx = fmaxf(mx, __shfl_xor(mx, 8, 32));
            const float mnew  = fmaxf(mrow[r], mx);
            const float alpha = __expf(mrow[r] - mnew);
            mrow[r] = mnew;
            float rs = 0.f;
#pragma unroll
            for (int ni = 0; ni < 4; ++ni) {
                const float p = __expf(sacc[ni][r] - mnew);
                sacc[ni][r] = p; rs += p;
            }
            rs += __shfl_xor(rs, 1, 32);
            rs += __shfl_xor(rs, 2, 32);
            rs += __shfl_xor(rs, 4, 32);
            rs += __shfl_xor(rs, 8, 32);
            lrow[r] = lrow[r] * alpha + rs;
#pragma unroll
            for (int di = 0; di < 4; ++di) acc_o[di][r] *= alpha;
        }

        // P (C-layout) -> LDS -> A-fragment layout.
#pragma unroll
        for (int ni = 0; ni < 4; ++ni)
#pragma unroll
            for (int r = 0; r < 8; ++r)
                Psh[wid][(r + half * 8) * LDK + ni * 16 + l16] = f2bf(sacc[ni][r]);
        __syncthreads();

        FragBF ap[2];
#pragma unroll
        for (int f = 0; f < 2; ++f)
#pragma unroll
            for (int j = 0; j < 8; ++j)
                ap[f].u[j] = *(const unsigned*)&Psh[wid][l16 * LDK + f * 32 + a_kk(j, half)];

        // O += P @ V
#pragma unroll
        for (int di = 0; di < 4; ++di) {
            FragBF bv0, bv1;
            const int dcol = di * 16 + l16;
#pragma unroll
            for (int j = 0; j < 8; ++j) {
                const int key = a_kk(j, half);
                bv0.u[j] = *(const unsigned*)&Vsh[dcol * LDK + key];
                bv1.u[j] = *(const unsigned*)&Vsh[dcol * LDK + 32 + key];
            }
            acc_o[di] = wmma_bf16(ap[0].v, bv0.v, acc_o[di]);
            acc_o[di] = wmma_bf16(ap[1].v, bv1.v, acc_o[di]);
        }
        __syncthreads();
    }

#pragma unroll
    for (int di = 0; di < 4; ++di)
#pragma unroll
        for (int r = 0; r < 8; ++r) {
            const int qr = qb * 64 + wid * 16 + r + half * 8;
            O[(size_t)(b * S + qr) * D + h * HD + di * 16 + l16] =
                f2bf(acc_o[di][r] / lrow[r]);
        }
}

// ---------------------------------------------------------------------------
// LayerNorm over D=1024; writes fp32 (residual path) + bf16 (GEMM operand).
// ---------------------------------------------------------------------------
__global__ __launch_bounds__(256) void layernorm_k(
    const float* __restrict__ in, const float* __restrict__ g,
    const float* __restrict__ be, float* __restrict__ out,
    ushort_t* __restrict__ outbf)
{
    constexpr int D = 1024;
    const int tid = threadIdx.x;
    const float* x = in + (size_t)blockIdx.x * D;
    float* y = out + (size_t)blockIdx.x * D;
    ushort_t* yb = outbf + (size_t)blockIdx.x * D;

    __shared__ float sb[256];
    float xv[4]; float s = 0.f;
#pragma unroll
    for (int i = 0; i < 4; ++i) { xv[i] = x[tid + i * 256]; s += xv[i]; }
    sb[tid] = s; __syncthreads();
    for (int off = 128; off > 0; off >>= 1) {
        if (tid < off) sb[tid] += sb[tid + off];
        __syncthreads();
    }
    const float mean = sb[0] * (1.f / D);
    __syncthreads();

    float v = 0.f;
#pragma unroll
    for (int i = 0; i < 4; ++i) { const float d = xv[i] - mean; v += d * d; }
    sb[tid] = v; __syncthreads();
    for (int off = 128; off > 0; off >>= 1) {
        if (tid < off) sb[tid] += sb[tid + off];
        __syncthreads();
    }
    const float inv = rsqrtf(sb[0] * (1.f / D) + 1e-6f);
#pragma unroll
    for (int i = 0; i < 4; ++i) {
        const int c = tid + i * 256;
        const float o = (xv[i] - mean) * inv * g[c] + be[c];
        y[c] = o;
        yb[c] = f2bf(o);
    }
}

// ---------------------------------------------------------------------------
// Host.  Workspace (bytes):
//   X f32[0,32M) Y f32[32,64) T f32[64,96) Xbf[96,112) Ybf[112,128)
//   Qbf[128,144) Kbf[144,160) Vbf[160,176) Cbf[176,192)  (Hbf aliases 128..192)
//   Wt bf16 region [192,216): Wqt Wkt Wvt Wpt (2MB each), W1t W2t (8MB each)
// ---------------------------------------------------------------------------
extern "C" void kernel_launch(void* const* d_in, const int* in_sizes, int n_in,
                              void* d_out, int out_size, void* d_ws, size_t ws_size,
                              hipStream_t stream) {
    constexpr int L = 4, Bb = 16, S = 512, D = 1024, F = 4096;
    constexpr int M = Bb * S;                    // 8192
    const size_t MD = (size_t)M * D;             // 8M elements
    const size_t MB = 1ull << 20;

    const float* hs = (const float*)d_in[0];
    const float* Wq = (const float*)d_in[1];  const float* bq = (const float*)d_in[2];
    const float* Wk = (const float*)d_in[3];  const float* bk = (const float*)d_in[4];
    const float* Wv = (const float*)d_in[5];  const float* bv = (const float*)d_in[6];
    const float* Wp = (const float*)d_in[7];  const float* bp = (const float*)d_in[8];
    const float* g1 = (const float*)d_in[9];  const float* be1 = (const float*)d_in[10];
    const float* W1 = (const float*)d_in[11]; const float* b1 = (const float*)d_in[12];
    const float* W2 = (const float*)d_in[13]; const float* b2 = (const float*)d_in[14];
    const float* g2 = (const float*)d_in[15]; const float* be2 = (const float*)d_in[16];

    char* ws = (char*)d_ws;
    float*    X   = (float*)(ws + 0 * MB);
    float*    Y   = (float*)(ws + 32 * MB);
    float*    T   = (float*)(ws + 64 * MB);
    ushort_t* Xbf = (ushort_t*)(ws + 96 * MB);
    ushort_t* Ybf = (ushort_t*)(ws + 112 * MB);
    ushort_t* Qbf = (ushort_t*)(ws + 128 * MB);
    ushort_t* Kbf = (ushort_t*)(ws + 144 * MB);
    ushort_t* Vbf = (ushort_t*)(ws + 160 * MB);
    ushort_t* Cbf = (ushort_t*)(ws + 176 * MB);
    ushort_t* Hbf = Qbf;                          // 64MB alias over Q..C
    ushort_t* Wqt = (ushort_t*)(ws + 192 * MB);
    ushort_t* Wkt = (ushort_t*)(ws + 194 * MB);
    ushort_t* Wvt = (ushort_t*)(ws + 196 * MB);
    ushort_t* Wpt = (ushort_t*)(ws + 198 * MB);
    ushort_t* W1t = (ushort_t*)(ws + 200 * MB);
    ushort_t* W2t = (ushort_t*)(ws + 208 * MB);

    hipMemcpyAsync(X, hs, MD * sizeof(float), hipMemcpyDeviceToDevice, stream);
    f32_to_bf16_k<<<(int)(MD / 1024), 256, 0, stream>>>(hs, Xbf);

    const dim3 blk(256);
    const dim3 gD(D / 128, M / 128);             // (8, 64)
    const dim3 gF(F / 128, M / 128);             // (32, 64)
    const dim3 gTdd(D / 32, D / 32);             // DxD transpose
    const dim3 gT1(F / 32, D / 32);              // W1 [D,F] -> [F,D]
    const dim3 gT2(D / 32, F / 32);              // W2 [F,D] -> [D,F]
    const int attnBlocks = Bb * 16 * (S / 64);   // 2048

    for (int l = 0; l < L; ++l) {
        const float* bql = bq + (size_t)l * D;  const float* bkl = bk + (size_t)l * D;
        const float* bvl = bv + (size_t)l * D;  const float* bpl = bp + (size_t)l * D;
        const float* b1l = b1 + (size_t)l * F;  const float* b2l = b2 + (size_t)l * D;
        const float* g1l = g1 + (size_t)l * D;  const float* be1l = be1 + (size_t)l * D;
        const float* g2l = g2 + (size_t)l * D;  const float* be2l = be2 + (size_t)l * D;

        transpose_f32_to_bf16<<<gTdd, blk, 0, stream>>>(Wq + (size_t)l * D * D, Wqt, D, D);
        transpose_f32_to_bf16<<<gTdd, blk, 0, stream>>>(Wk + (size_t)l * D * D, Wkt, D, D);
        transpose_f32_to_bf16<<<gTdd, blk, 0, stream>>>(Wv + (size_t)l * D * D, Wvt, D, D);
        transpose_f32_to_bf16<<<gTdd, blk, 0, stream>>>(Wp + (size_t)l * D * D, Wpt, D, D);
        transpose_f32_to_bf16<<<gT1,  blk, 0, stream>>>(W1 + (size_t)l * D * F, W1t, D, F);
        transpose_f32_to_bf16<<<gT2,  blk, 0, stream>>>(W2 + (size_t)l * F * D, W2t, F, D);

        gemm_bf16_wmma<0><<<gD, blk, 0, stream>>>(Xbf, Wqt, bql, nullptr, Qbf, M, D, D);
        gemm_bf16_wmma<0><<<gD, blk, 0, stream>>>(Xbf, Wkt, bkl, nullptr, Kbf, M, D, D);
        gemm_bf16_wmma<0><<<gD, blk, 0, stream>>>(Xbf, Wvt, bvl, nullptr, Vbf, M, D, D);

        attn_wmma<<<attnBlocks, dim3(128), 0, stream>>>(Qbf, Kbf, Vbf, Cbf);

        // pre-LN1 = ctx @ Wp + bp + x
        gemm_bf16_wmma<2><<<gD, blk, 0, stream>>>(Cbf, Wpt, bpl, X, T, M, D, D);
        layernorm_k<<<M, blk, 0, stream>>>(T, g1l, be1l, Y, Ybf);

        // FFN: h = relu(y @ W1 + b1) -> bf16  (Hbf aliases dead Q..C region)
        gemm_bf16_wmma<1><<<gF, blk, 0, stream>>>(Ybf, W1t, b1l, nullptr, Hbf, M, F, D);
        // pre-LN2 = h @ W2 + b2 + y
        gemm_bf16_wmma<2><<<gD, blk, 0, stream>>>(Hbf, W2t, b2l, Y, T, M, D, F);

        float* xo = (l == L - 1) ? (float*)d_out : X;
        layernorm_k<<<M, blk, 0, stream>>>(T, g2l, be2l, xo, Xbf);
    }
}
